// EfficientDETRDecoder_48438641164710
// MI455X (gfx1250) — matrix operand order repaired and checked by reading, not compiled, not run on hardware
//
#include <hip/hip_runtime.h>
#include <hip/hip_bf16.h>
#include <stdint.h>
#include <math.h>

typedef __attribute__((ext_vector_type(16))) _Float16 v16h;
typedef __attribute__((ext_vector_type(8)))  float    v8f;
typedef __attribute__((ext_vector_type(4)))  unsigned int u32x4;
typedef __attribute__((ext_vector_type(8)))  int      i32x8;
typedef __attribute__((ext_vector_type(4)))  int      i32x4;

#if defined(__has_builtin)
#if __has_builtin(__builtin_amdgcn_tensor_load_to_lds) && \
    __has_builtin(__builtin_amdgcn_s_wait_tensorcnt)
#define ATHENA_TDM 1
#endif
#endif

namespace {

constexpr int Bn = 16, Cc = 256, Hh = 100, Ww = 100, HW = Hh * Ww;
constexpr int NQ = 300, NHh = 8, NPp = 4, HD = Cc / NHh;   // HD = 32
constexpr int ROWS = Bn * NQ;        // 4800  (divisible by 64)
constexpr int MBIG = Bn * HW;        // 160000 (divisible by 64)

union Frag16 { v16h v; uint32_t u[8]; };

// ---------------------------------------------------------------------------
// feat_seq = (feat_map + sine positional encoding) reshaped to [B, HW, C], f16
// ---------------------------------------------------------------------------
__global__ void pos_feat_kernel(const float* __restrict__ fm,
                                _Float16* __restrict__ out, int total) {
  int idx = blockIdx.x * blockDim.x + threadIdx.x;
  if (idx >= total) return;
  int c = idx % Cc;
  int s = (idx / Cc) % HW;
  int b = idx / (Cc * HW);
  int yy = s / Ww, xx = s % Ww;
  int i = (c < 128) ? c : (c - 128);
  float coord = (c < 128) ? ((float)yy / (float)Hh) : ((float)xx / (float)Ww);
  float t = __powf(10000.0f, (2.0f * (float)(i / 2)) / 128.0f);
  float arg = coord / t;
  float pos = (i & 1) ? __cosf(arg) : __sinf(arg);
  float v = fm[((size_t)b * Cc + c) * HW + s] + pos;
  out[idx] = (_Float16)v;
}

__global__ void broadcast_q_kernel(const float* __restrict__ qe,
                                   float* __restrict__ q, int total) {
  int idx = blockIdx.x * blockDim.x + threadIdx.x;
  if (idx >= total) return;
  q[idx] = qe[idx % (NQ * Cc)];
}

__global__ void f32_to_f16_kernel(const float* __restrict__ in,
                                  _Float16* __restrict__ out, int n) {
  int idx = blockIdx.x * blockDim.x + threadIdx.x;
  if (idx >= n) return;
  out[idx] = (_Float16)in[idx];
}

__global__ void add_inplace_kernel(float* __restrict__ a,
                                   const float* __restrict__ b, int n) {
  int idx = blockIdx.x * blockDim.x + threadIdx.x;
  if (idx >= n) return;
  a[idx] += b[idx];
}

// ---------------------------------------------------------------------------
// LayerNorm over C=256, one row per 256-thread block
// ---------------------------------------------------------------------------
__global__ void layernorm_kernel(const float* __restrict__ x,
                                 const float* __restrict__ sc,
                                 const float* __restrict__ bi,
                                 float* __restrict__ out) {
  __shared__ float red[256];
  int row = blockIdx.x, t = threadIdx.x;
  float v = x[(size_t)row * Cc + t];
  red[t] = v;
  __syncthreads();
  for (int o = 128; o > 0; o >>= 1) {
    if (t < o) red[t] += red[t + o];
    __syncthreads();
  }
  float mean = red[0] / (float)Cc;
  __syncthreads();
  float dv = v - mean;
  red[t] = dv * dv;
  __syncthreads();
  for (int o = 128; o > 0; o >>= 1) {
    if (t < o) red[t] += red[t + o];
    __syncthreads();
  }
  float var = red[0] / (float)Cc;
  out[(size_t)row * Cc + t] = dv * rsqrtf(var + 1e-5f) * sc[t] + bi[t];
}

// ---------------------------------------------------------------------------
// Generic WMMA GEMM: Out[m,n] = act( sum_k A[m,k] * W[n,k] + bias[n] )
//   A: [M,K] f16 row-major,  W: [N,K] f16 row-major (so B = W^T)
//   block = (32,4): 4 waves cover a 64x16 tile.
//   W tile staged into LDS by the Tensor Data Mover (TDM) when available.
//   Inner loop is a 2-deep software pipeline: load frags k+1, WMMA on k.
//   Requires: M % 64 == 0, N % 16 == 0, K in {256, 512}.
// ---------------------------------------------------------------------------
template <typename OutT, bool RELU, int K>
__global__ void gemm16_kernel(const _Float16* __restrict__ A,
                              const _Float16* __restrict__ Wt,
                              const float* __restrict__ bias,
                              OutT* __restrict__ Out, int M, int N) {
  __shared__ _Float16 wtile[16 * K];
  const int nt = blockIdx.x;
  const int lane = threadIdx.x;            // 0..31
  const int wv = threadIdx.y;              // 0..3

#if defined(ATHENA_TDM)
  // -- Stage the 16xK f16 W tile into LDS with one TDM descriptor ----------
  if (wv == 0) {
    uint64_t ga = (uint64_t)(uintptr_t)(Wt + (size_t)nt * 16 * K);
    // LDS aperture: generic LDS address truncated to 32 bits == LDS offset.
    uint32_t la = (uint32_t)(uintptr_t)&wtile[0];
    u32x4 g0;
    g0[0] = 1u;                                   // count=1, no gather
    g0[1] = la;                                   // lds_addr
    g0[2] = (uint32_t)(ga & 0xFFFFFFFFu);         // global_addr[31:0]
    g0[3] = (uint32_t)((ga >> 32) & 0x1FFFFFFu)   // global_addr[56:32]
            | (2u << 30);                         // type = 2 ("image")
    i32x8 g1;
    g1[0] = (int)(1u << 16);                      // data_size = 1 (2 bytes)
    g1[1] = (int)(((unsigned)K & 0xFFFFu) << 16); // tensor_dim0[15:0] = K
    g1[2] = (int)(((unsigned)K >> 16) | (16u << 16)); // dim0 hi | tensor_dim1=16
    g1[3] = (int)(((unsigned)K & 0xFFFFu) << 16); // dim1 hi=0 | tile_dim0 = K
    g1[4] = 16;                                   // tile_dim1 = 16, tile_dim2 = 0
    g1[5] = K;                                    // tensor_dim0_stride[31:0] = K
    g1[6] = (int)(((unsigned)K & 0xFFFFu) << 16); // stride0 hi=0 | stride1 lo = K
    g1[7] = (int)((unsigned)K >> 16);             // stride1 hi
    i32x4 z4 = {0, 0, 0, 0};
#if __clang_major__ >= 23
    i32x8 z8 = {0, 0, 0, 0, 0, 0, 0, 0};
    __builtin_amdgcn_tensor_load_to_lds(g0, g1, z4, z4, z8, 0);
#else
    __builtin_amdgcn_tensor_load_to_lds(g0, g1, z4, z4, 0);
#endif
    __builtin_amdgcn_s_wait_tensorcnt(0);
  }
  __syncthreads();
#else
  // -- Fallback: cooperative dword staging ---------------------------------
  {
    const int tid = wv * 32 + lane;
    const int totalHalf = 16 * K;
    for (int idx = tid * 2; idx < totalHalf; idx += 256) {
      int rw = idx / K, col = idx % K;
      *(uint32_t*)&wtile[idx] =
          *(const uint32_t*)&Wt[(size_t)(nt * 16 + rw) * K + col];
    }
    __syncthreads();
  }
#endif

  const int mt = blockIdx.y * 4 + wv;
  const int g = lane >> 4, r = lane & 15;
  const _Float16* arow = A + (size_t)(mt * 16 + r) * K;
  const _Float16* wrow = wtile + r * K;

  constexpr int NKC = K / 32;
  Frag16 fa[2], fb[2];

  auto loadFrag = [&](int kc, Frag16& xa, Frag16& xb) {
    const int k0 = kc * 32;
#pragma unroll
    for (int i = 0; i < 8; ++i) {
      // A fragment: 16-bit A 16x32 layout (ISA 7.12.2)
      int ka = ((i & 4) ? 16 : 0) + 8 * g + (i & 3) * 2;
      xa.u[i] = *(const uint32_t*)(arow + k0 + ka);
      // B fragment: same k order as A (lanes 0-15: K=0..7,16..23; 16-31: 8..15,24..31)
      int kb = ((i & 4) ? 16 : 0) + 8 * g + (i & 3) * 2;
      xb.u[i] = *(const uint32_t*)(wrow + k0 + kb);
    }
  };

  v8f acc = {};
  loadFrag(0, fa[0], fb[0]);
#pragma unroll
  for (int kc = 0; kc < NKC; ++kc) {
    const int cur = kc & 1;
    if (kc + 1 < NKC) loadFrag(kc + 1, fa[cur ^ 1], fb[cur ^ 1]);
    acc = __builtin_amdgcn_wmma_f32_16x16x32_f16(
        false, fa[cur].v, false, fb[cur].v, (short)0, acc, false, false);
  }

#pragma unroll
  for (int rr = 0; rr < 8; ++rr) {
    int m = mt * 16 + rr + 8 * g;
    int n = nt * 16 + r;
    float val = acc[rr] + (bias ? bias[n] : 0.0f);
    if (RELU) val = fmaxf(val, 0.0f);
    Out[(size_t)m * N + n] = (OutT)val;
  }
}

// ---------------------------------------------------------------------------
// MHA core: online-softmax attention per (b, h, q) thread. qkv: [B*NQ, 768].
// ---------------------------------------------------------------------------
__global__ void mha_attn_kernel(const float* __restrict__ qkv,
                                float* __restrict__ o) {
  int gid = blockIdx.x * blockDim.x + threadIdx.x;
  if (gid >= Bn * NHh * NQ) return;
  int qi = gid % NQ;
  int h = (gid / NQ) % NHh;
  int b = gid / (NQ * NHh);
  const float* base = qkv + (size_t)b * NQ * (3 * Cc);
  float qv[HD];
#pragma unroll
  for (int d = 0; d < HD; ++d) qv[d] = base[(size_t)qi * (3 * Cc) + h * HD + d];
  float m = -3.0e38f, l = 0.0f, acc[HD];
#pragma unroll
  for (int d = 0; d < HD; ++d) acc[d] = 0.0f;
  for (int k = 0; k < NQ; ++k) {
    const float* kp = base + (size_t)k * (3 * Cc) + Cc + h * HD;
    float s = 0.0f;
#pragma unroll
    for (int d = 0; d < HD; ++d) s += qv[d] * kp[d];
    s *= 0.17677669529663687f;  // 1/sqrt(32)
    float mn = fmaxf(m, s);
    float corr = __expf(m - mn);
    float p = __expf(s - mn);
    const float* vp = base + (size_t)k * (3 * Cc) + 2 * Cc + h * HD;
    l = l * corr + p;
#pragma unroll
    for (int d = 0; d < HD; ++d) acc[d] = acc[d] * corr + p * vp[d];
    m = mn;
  }
  float inv = 1.0f / l;
#pragma unroll
  for (int d = 0; d < HD; ++d)
    o[((size_t)b * NQ + qi) * Cc + h * HD + d] = acc[d] * inv;
}

// ---------------------------------------------------------------------------
// Deformable-attention bilinear sampling from f16 value buffer v[B, HW, C].
// block = 256 (= NH * HD), grid = (NQ, B)
// ---------------------------------------------------------------------------
__device__ __forceinline__ float tap(const _Float16* vb, int xc, int yc) {
  if (xc < 0 || xc >= Ww || yc < 0 || yc >= Hh) return 0.0f;
  return (float)vb[((size_t)yc * Ww + xc) * Cc];
}

__global__ void deform_sample_kernel(const float* __restrict__ doff,
                                     const float* __restrict__ awl,
                                     const float* __restrict__ ref,
                                     const _Float16* __restrict__ vh,
                                     float* __restrict__ samp) {
  int q = blockIdx.x, b = blockIdx.y;
  int h = threadIdx.x >> 5, d = threadIdx.x & 31;
  int row = b * NQ + q;

  float wts[NPp];
  float mx = -3.0e38f;
#pragma unroll
  for (int p = 0; p < NPp; ++p) {
    wts[p] = awl[(size_t)row * (NHh * NPp) + h * NPp + p];
    mx = fmaxf(mx, wts[p]);
  }
  float sum = 0.0f;
#pragma unroll
  for (int p = 0; p < NPp; ++p) {
    wts[p] = __expf(wts[p] - mx);
    sum += wts[p];
  }
  float isum = 1.0f / sum;
  float rx = ref[row * 2 + 0], ry = ref[row * 2 + 1];
  const _Float16* vb = vh + (size_t)b * HW * Cc + h * HD + d;

  float acc = 0.0f;
#pragma unroll
  for (int p = 0; p < NPp; ++p) {
    float ox = doff[(size_t)row * (NHh * NPp * 2) + (h * NPp + p) * 2 + 0] *
               (0.1f / (float)Ww);
    float oy = doff[(size_t)row * (NHh * NPp * 2) + (h * NPp + p) * 2 + 1] *
               (0.1f / (float)Hh);
    float lx = fminf(fmaxf(rx + ox, 0.0f), 1.0f);
    float ly = fminf(fmaxf(ry + oy, 0.0f), 1.0f);
    float xf = lx * (float)Ww - 0.5f;
    float yf = ly * (float)Hh - 0.5f;
    float x0f = floorf(xf), y0f = floorf(yf);
    int x0 = (int)x0f, y0 = (int)y0f;
    float wx1 = xf - x0f, wx0 = 1.0f - wx1;
    float wy1 = yf - y0f, wy0 = 1.0f - wy1;
    float s = tap(vb, x0, y0) * wx0 * wy0 + tap(vb, x0 + 1, y0) * wx1 * wy0 +
              tap(vb, x0, y0 + 1) * wx0 * wy1 +
              tap(vb, x0 + 1, y0 + 1) * wx1 * wy1;
    acc += wts[p] * isum * s;
  }
  samp[(size_t)row * Cc + h * HD + d] = acc;
}

// ---------------------------------------------------------------------------
// Small scalar heads (N <= 4): Out[row,n] = act(X[row,:] . W[n,:] + b[n])
// ---------------------------------------------------------------------------
template <int N, bool SIG>
__global__ void small_head_kernel(const float* __restrict__ X,
                                  const float* __restrict__ Wt,
                                  const float* __restrict__ bias,
                                  float* __restrict__ out, int rows, int K) {
  int row = blockIdx.x * blockDim.x + threadIdx.x;
  if (row >= rows) return;
#pragma unroll
  for (int n = 0; n < N; ++n) {
    float s = bias[n];
    for (int k = 0; k < K; ++k) s += X[(size_t)row * K + k] * Wt[n * K + k];
    if (SIG) s = 1.0f / (1.0f + __expf(-s));
    out[(size_t)row * N + n] = s;
  }
}

__global__ void ref_update_kernel(const float* __restrict__ bb,
                                  float* __restrict__ ref) {
  int row = blockIdx.x * blockDim.x + threadIdx.x;
  if (row >= ROWS) return;
  ref[row * 2 + 0] = 0.5f * (bb[row * 4 + 0] + bb[row * 4 + 2]);
  ref[row * 2 + 1] = 0.5f * (bb[row * 4 + 1] + bb[row * 4 + 3]);
}

// d_out = [bb1 (19200) | sc1 (4800) | bb0 (19200) | sc0 (4800)]
__global__ void pack_kernel(const float* __restrict__ bbL,
                            const float* __restrict__ scL,
                            const float* __restrict__ bb0,
                            const float* __restrict__ sc0,
                            float* __restrict__ out) {
  int idx = blockIdx.x * blockDim.x + threadIdx.x;
  if (idx >= ROWS * 10) return;
  if (idx < ROWS * 4) out[idx] = bbL[idx];
  else if (idx < ROWS * 5) out[idx] = scL[idx - ROWS * 4];
  else if (idx < ROWS * 9) out[idx] = bb0[idx - ROWS * 5];
  else out[idx] = sc0[idx - ROWS * 9];
}

}  // namespace

// ===========================================================================
// Host orchestration
// ===========================================================================
extern "C" void kernel_launch(void* const* d_in, const int* in_sizes, int n_in,
                              void* d_out, int out_size, void* d_ws,
                              size_t ws_size, hipStream_t stream) {
  (void)in_sizes; (void)n_in; (void)out_size; (void)ws_size;
  const float* feat_map    = (const float*)d_in[0];
  const float* query_embed = (const float*)d_in[1];
  const float* ref_w       = (const float*)d_in[2];
  const float* ref_b       = (const float*)d_in[3];
  // Per-layer params, dict order (28 per layer):
  // 0 n1_s 1 n1_b 2 n2_s 3 n2_b 4 n3_s 5 n3_b 6 sa_in_w 7 sa_in_b 8 sa_out_w
  // 9 sa_out_b 10 off_w 11 off_b 12 aw_w 13 aw_b 14 vp_w 15 vp_b 16 op_w
  // 17 op_b 18 ffn_w1 19 ffn_b1 20 ffn_w2 21 ffn_b2 22 bb_w1 23 bb_b1
  // 24 bb_w2 25 bb_b2 26 os_w 27 os_b
  auto LP = [&](int li, int j) -> const float* {
    return (const float*)d_in[4 + li * 28 + j];
  };

  char* wbase = (char*)d_ws;
  size_t wsoff = 0;
  auto alloc = [&](size_t bytes) -> void* {
    void* p = wbase + wsoff;
    wsoff = (wsoff + bytes + 255) & ~(size_t)255;
    return p;
  };

  _Float16* feat_h = (_Float16*)alloc((size_t)MBIG * Cc * 2);
  _Float16* v_h    = (_Float16*)alloc((size_t)MBIG * Cc * 2);
  _Float16* wh     = (_Float16*)alloc((size_t)768 * 512 * 2);
  _Float16* ah     = (_Float16*)alloc((size_t)ROWS * 512 * 2);
  float* q    = (float*)alloc((size_t)ROWS * Cc * 4);
  float* qn   = (float*)alloc((size_t)ROWS * Cc * 4);
  float* t0   = (float*)alloc((size_t)ROWS * 768 * 4);
  float* t1   = (float*)alloc((size_t)ROWS * Cc * 4);
  float* offb = (float*)alloc((size_t)ROWS * 64 * 4);
  float* awb  = (float*)alloc((size_t)ROWS * 32 * 4);
  float* refb = (float*)alloc((size_t)ROWS * 2 * 4);
  float* bbuf0 = (float*)alloc((size_t)ROWS * 4 * 4);
  float* bbuf1 = (float*)alloc((size_t)ROWS * 4 * 4);
  float* scb0  = (float*)alloc((size_t)ROWS * 4);
  float* scb1  = (float*)alloc((size_t)ROWS * 4);

  auto conv = [&](const float* in, _Float16* outp, int n) {
    f32_to_f16_kernel<<<(n + 255) / 256, 256, 0, stream>>>(in, outp, n);
  };
  auto add = [&](float* a, const float* b, int n) {
    add_inplace_kernel<<<(n + 255) / 256, 256, 0, stream>>>(a, b, n);
  };
  auto ln = [&](const float* x, const float* s, const float* b, float* o) {
    layernorm_kernel<<<ROWS, 256, 0, stream>>>(x, s, b, o);
  };
  auto gemmF = [&](const _Float16* A, const float* w32, const float* bias,
                   float* Out, int M, int N, int K, bool relu) {
    conv(w32, wh, N * K);
    dim3 g(N / 16, M / 64), blk(32, 4);
    if (K == 512) {
      gemm16_kernel<float, false, 512><<<g, blk, 0, stream>>>(A, wh, bias, Out, M, N);
    } else if (relu) {
      gemm16_kernel<float, true, 256><<<g, blk, 0, stream>>>(A, wh, bias, Out, M, N);
    } else {
      gemm16_kernel<float, false, 256><<<g, blk, 0, stream>>>(A, wh, bias, Out, M, N);
    }
  };
  auto gemmH = [&](const _Float16* A, const float* w32, const float* bias,
                   _Float16* Out, int M, int N) {
    conv(w32, wh, N * 256);
    dim3 g(N / 16, M / 64), blk(32, 4);
    gemm16_kernel<_Float16, false, 256><<<g, blk, 0, stream>>>(A, wh, bias, Out, M, N);
  };

  // ---- preamble --------------------------------------------------------
  {
    int total = Bn * HW * Cc;
    pos_feat_kernel<<<(total + 255) / 256, 256, 0, stream>>>(feat_map, feat_h, total);
  }
  {
    int total = ROWS * Cc;
    broadcast_q_kernel<<<(total + 255) / 256, 256, 0, stream>>>(query_embed, q, total);
  }
  small_head_kernel<2, true><<<(ROWS + 127) / 128, 128, 0, stream>>>(
      q, ref_w, ref_b, refb, ROWS, Cc);

  // ---- decoder layers --------------------------------------------------
  for (int li = 0; li < 2; ++li) {
    float* bb = (li == 0) ? bbuf0 : bbuf1;
    float* sc = (li == 0) ? scb0 : scb1;

    // --- self attention ---
    ln(q, LP(li, 0), LP(li, 1), qn);
    conv(qn, ah, ROWS * Cc);
    gemmF(ah, LP(li, 6), LP(li, 7), t0, ROWS, 3 * Cc, Cc, false);   // qkv
    {
      int total = Bn * NHh * NQ;
      mha_attn_kernel<<<(total + 127) / 128, 128, 0, stream>>>(t0, t1);
    }
    conv(t1, ah, ROWS * Cc);
    gemmF(ah, LP(li, 8), LP(li, 9), t0, ROWS, Cc, Cc, false);       // out proj
    add(q, t0, ROWS * Cc);

    // --- deformable attention ---
    ln(q, LP(li, 2), LP(li, 3), qn);
    conv(qn, ah, ROWS * Cc);
    gemmF(ah, LP(li, 10), LP(li, 11), offb, ROWS, 64, Cc, false);   // offsets
    gemmF(ah, LP(li, 12), LP(li, 13), awb, ROWS, 32, Cc, false);    // attn wts
    gemmH(feat_h, LP(li, 14), LP(li, 15), v_h, MBIG, Cc);           // value proj (big)
    deform_sample_kernel<<<dim3(NQ, Bn), 256, 0, stream>>>(offb, awb, refb, v_h, t1);
    conv(t1, ah, ROWS * Cc);
    gemmF(ah, LP(li, 16), LP(li, 17), t0, ROWS, Cc, Cc, false);     // op proj
    add(q, t0, ROWS * Cc);

    // --- FFN ---
    ln(q, LP(li, 4), LP(li, 5), qn);
    conv(qn, ah, ROWS * Cc);
    gemmF(ah, LP(li, 18), LP(li, 19), t0, ROWS, 2 * Cc, Cc, true);  // ffn1 + relu
    conv(t0, ah, ROWS * 2 * Cc);
    gemmF(ah, LP(li, 20), LP(li, 21), t1, ROWS, Cc, 2 * Cc, false); // ffn2
    add(q, t1, ROWS * Cc);

    // --- heads ---
    conv(q, ah, ROWS * Cc);
    gemmF(ah, LP(li, 22), LP(li, 23), t0, ROWS, Cc, Cc, true);      // bb hidden
    small_head_kernel<4, true><<<(ROWS + 127) / 128, 128, 0, stream>>>(
        t0, LP(li, 24), LP(li, 25), bb, ROWS, Cc);
    small_head_kernel<1, false><<<(ROWS + 127) / 128, 128, 0, stream>>>(
        q, LP(li, 26), LP(li, 27), sc, ROWS, Cc);
    if (li == 0)
      ref_update_kernel<<<(ROWS + 127) / 128, 128, 0, stream>>>(bb, refb);
  }

  // ---- pack outputs ----------------------------------------------------
  {
    int total = ROWS * 10;
    pack_kernel<<<(total + 255) / 256, 256, 0, stream>>>(
        bbuf1, scb1, bbuf0, scb0, (float*)d_out);
  }
}
